// MultiHeadAttention_78726750536399
// MI455X (gfx1250) — matrix operand order repaired and checked by reading, not compiled, not run on hardware
//
#include <hip/hip_runtime.h>
#include <hip/hip_bf16.h>
#include <math.h>

// MI455X / gfx1250: wave32, WMMA 16x16x32 bf16 path.
typedef __attribute__((ext_vector_type(16))) __bf16 v16bf;
typedef __attribute__((ext_vector_type(8)))  __bf16 v8bf;
typedef __attribute__((ext_vector_type(8)))  float  v8f;
typedef __attribute__((ext_vector_type(4)))  float  v4f;

// Pointer types for the gfx1250 async global->LDS builtin:
// param0: int4-vector pointer in AS(1) (global), param1: int4-vector pointer in AS(3) (LDS).
typedef int i4v __attribute__((vector_size(16)));
typedef i4v __attribute__((address_space(1))) i4v_g;
typedef i4v __attribute__((address_space(3))) i4v_l;

#define BB 4
#define TT 2048
#define CC 1024
#define HH 16
#define HD 64

#if defined(__gfx1250__) && __has_builtin(__builtin_amdgcn_global_load_async_to_lds_b128) && \
    __has_builtin(__builtin_amdgcn_s_wait_asynccnt)
#define USE_ASYNC_LDS 1
#else
#define USE_ASYNC_LDS 0
#endif

__device__ __forceinline__ v8f wmma_bf16(v16bf a, v16bf b, v8f c) {
  // (neg_a, A, neg_b, B, c_mod, C, reuse_a, reuse_b)
  return __builtin_amdgcn_wmma_f32_16x16x32_bf16(false, a, false, b, (short)0, c, false, false);
}

// A-fragment (16x32 bf16): lane = row M (lane&15); half-wave selects K sub-block.
// lane holds K = {kb..kb+15}, kb = (lane>>4)*16. Caller pre-offsets
// p by row*ld + half*16 (+k0); we read 8 elems @ +0 and 8 @ +8.
__device__ __forceinline__ v16bf a_frag_bf16(const __bf16* p) {
  v8bf x = *(const v8bf*)p;
  v8bf y = *(const v8bf*)(p + 8);
  v16bf a;
#pragma unroll
  for (int i = 0; i < 8; i++) { a[i] = x[i]; a[i + 8] = y[i]; }
  return a;
}

__device__ __forceinline__ v16bf a_frag_f32(const float* p) {
  v4f f0 = *(const v4f*)(p);
  v4f f1 = *(const v4f*)(p + 4);
  v4f f2 = *(const v4f*)(p + 8);
  v4f f3 = *(const v4f*)(p + 12);
  v16bf a;
#pragma unroll
  for (int i = 0; i < 4; i++) {
    a[i]      = (__bf16)f0[i];
    a[i + 4]  = (__bf16)f1[i];
    a[i + 8]  = (__bf16)f2[i];
    a[i + 12] = (__bf16)f3[i];
  }
  return a;
}

// ---------------- weight f32 -> bf16 conversion ----------------
__global__ void cvt_f32_bf16(const float* __restrict__ in, __bf16* __restrict__ out, int n) {
  for (int i = blockIdx.x * blockDim.x + threadIdx.x; i < n; i += gridDim.x * blockDim.x)
    out[i] = (__bf16)in[i];
}

// ---------------- fused QKV projection: Y = X @ W^T -------------
// Per wave: 32x64 output tile (8 accumulators); manual ping-pong double buffer
// (k-loop unrolled by 2) so buffer rotation is pure register renaming.
// Stores q,k as [B,H,T,64]; v transposed [B,H,64,T].
__global__ void __launch_bounds__(256)
qkv_gemm(const float* __restrict__ xq, const float* __restrict__ xk, const float* __restrict__ xv,
         const __bf16* __restrict__ wq, const __bf16* __restrict__ wk, const __bf16* __restrict__ wv,
         __bf16* __restrict__ qh, __bf16* __restrict__ kh, __bf16* __restrict__ vT) {
  int wid  = blockIdx.x * 8 + (threadIdx.x >> 5);
  int mat  = wid >> 12;          // 0=q,1=k,2=v  (4096 wave-tiles per matrix)
  int rem  = wid & 4095;
  int mt   = rem >> 4;           // 256 m-tiles of 32 rows
  int ng   = rem & 15;           // 16 groups of 64 columns
  int lane = threadIdx.x & 31;
  int n0   = lane & 15;
  int half = lane >> 4;

  const float*  X = (mat == 0) ? xq : (mat == 1) ? xk : xv;
  const __bf16* W = (mat == 0) ? wq : (mat == 1) ? wk : wv;

  int base_m = mt * 32, base_n = ng * 64;
  const float*  pa0 = X + (size_t)(base_m + n0) * CC + half * 16;       // rows 0..15
  const float*  pa1 = X + (size_t)(base_m + 16 + n0) * CC + half * 16;  // rows 16..31
  const __bf16* pw  = W + (size_t)(base_n + n0) * CC + half * 16;

  v8f acc[8] = {};
  v16bf A0[2], A1[2], Bt[2][4];

  // prime buffer 0 with k0 = 0
  A0[0] = a_frag_f32(pa0);
  A1[0] = a_frag_f32(pa1);
#pragma unroll
  for (int j = 0; j < 4; j++) Bt[0][j] = *(const v16bf*)(pw + (size_t)j * 16 * CC);

  for (int k0 = 0; k0 < CC; k0 += 64) {
    // preload buffer 1 with k0+32 (always in range: k0 <= 960)
    __builtin_prefetch(pa0 + k0 + 192, 0, 1);   // global_prefetch_b8
    A0[1] = a_frag_f32(pa0 + k0 + 32);
    A1[1] = a_frag_f32(pa1 + k0 + 32);
#pragma unroll
    for (int j = 0; j < 4; j++)
      Bt[1][j] = *(const v16bf*)(pw + (size_t)j * 16 * CC + k0 + 32);

    // compute with buffer 0 (k0)
#pragma unroll
    for (int j = 0; j < 4; j++) {
      acc[j]     = wmma_bf16(A0[0], Bt[0][j], acc[j]);
      acc[j + 4] = wmma_bf16(A1[0], Bt[0][j], acc[j + 4]);
    }

    // preload buffer 0 with k0+64
    if (k0 + 64 < CC) {
      A0[0] = a_frag_f32(pa0 + k0 + 64);
      A1[0] = a_frag_f32(pa1 + k0 + 64);
#pragma unroll
      for (int j = 0; j < 4; j++)
        Bt[0][j] = *(const v16bf*)(pw + (size_t)j * 16 * CC + k0 + 64);
    }

    // compute with buffer 1 (k0+32)
#pragma unroll
    for (int j = 0; j < 4; j++) {
      acc[j]     = wmma_bf16(A0[1], Bt[1][j], acc[j]);
      acc[j + 4] = wmma_bf16(A1[1], Bt[1][j], acc[j + 4]);
    }
  }

#pragma unroll
  for (int s2 = 0; s2 < 2; s2++) {
#pragma unroll
    for (int j = 0; j < 4; j++) {
#pragma unroll
      for (int r = 0; r < 8; r++) {
        int m = base_m + s2 * 16 + r + half * 8;
        int d = base_n + j * 16 + n0;
        int b = m >> 11, t = m & (TT - 1);
        int h = d >> 6,  dd = d & (HD - 1);
        __bf16 val = (__bf16)acc[s2 * 4 + j][r];
        if (mat == 0)
          qh[(((size_t)(b * HH + h)) * TT + t) * HD + dd] = val;
        else if (mat == 1)
          kh[(((size_t)(b * HH + h)) * TT + t) * HD + dd] = val;
        else
          vT[(((size_t)(b * HH + h)) * HD + dd) * TT + t] = val;
      }
    }
  }
}

// ---------------- attention core --------------------------------
// One block = one (b,h) x 16 query rows. S[16][2048] f32 in dynamic LDS.
__global__ void __launch_bounds__(256)
attn_kernel(const __bf16* __restrict__ qh, const __bf16* __restrict__ kh,
            const __bf16* __restrict__ vT, const int* __restrict__ mask,
            __bf16* __restrict__ ao) {
  extern __shared__ float smem[];
  float*  S       = smem;                           // 16 * 2048 f32
  float*  scratch = smem + 16 * TT;                 // 16 * 64 f32
  __bf16* qtile   = (__bf16*)(scratch + 16 * 64);   // 16 * 64 bf16

  int bh = blockIdx.x >> 7;         // b*H + h
  int qt = blockIdx.x & 127;
  int t0 = qt * 16;
  int tid = threadIdx.x;
  int w = tid >> 5, lane = tid & 31;
  int n0 = lane & 15, half = lane >> 4;

  const __bf16* qp = qh + (size_t)bh * TT * HD;
  const __bf16* kp = kh + (size_t)bh * TT * HD;
  const __bf16* vp = vT + (size_t)bh * HD * TT;

#if USE_ASYNC_LDS
  // Stage the 2 KB Q tile into LDS with gfx1250 async copies (128 lanes x 16B).
  if (tid < 128) {
    int row = tid >> 3, seg = tid & 7;
    const __bf16* g = qp + (size_t)(t0 + row) * HD + seg * 8;
    __bf16*       l = qtile + row * HD + seg * 8;
    __builtin_amdgcn_global_load_async_to_lds_b128(
        (i4v_g*)(uintptr_t)g,
        (i4v_l*)(unsigned)(uintptr_t)l, 0, 0);
  }
  __builtin_amdgcn_s_wait_asynccnt(0);
  __syncthreads();
  const __bf16* qrow = qtile + n0 * HD + half * 16;
#else
  const __bf16* qrow = qp + (size_t)(t0 + n0) * HD + half * 16;
#endif

  // Q A-fragments (K = head_dim = 64 -> two 16x32 fragments), reused for all key tiles.
  v16bf A0 = a_frag_bf16(qrow);
  v16bf A1 = a_frag_bf16(qrow + 32);

  // Phase 1: S = (Q K^T) * C^-0.5, zero-fill masked entries (reference quirk).
  // Unrolled so K-fragment double buffering is register renaming, not copies.
#pragma unroll 4
  for (int ct = w * 16; ct < w * 16 + 16; ct++) {
    int kcol = ct * 16 + n0;
    const __bf16* kb = kp + (size_t)kcol * HD + half * 16;
    v16bf B0 = *(const v16bf*)kb;
    v16bf B1 = *(const v16bf*)(kb + 32);
    v8f s = {};
    s = wmma_bf16(A0, B0, s);
    s = wmma_bf16(A1, B1, s);
#pragma unroll
    for (int r = 0; r < 8; r++) {
      int m = r + half * 8;
      int qpos = t0 + m;
      int mv = mask[(size_t)qpos * TT + kcol];
      S[m * TT + kcol] = (mv == 0) ? 0.0f : s[r] * 0.03125f;  // 1024^-0.5 = 1/32
    }
  }
  __syncthreads();

  // Phase 2: row-wise softmax (16 threads per row, 16-lane shuffle reductions).
  {
    int row = tid >> 4, sub = tid & 15;
    float* Sr = S + row * TT;
    float mx = -3.0e38f;
    for (int i = sub; i < TT; i += 16) mx = fmaxf(mx, Sr[i]);
#pragma unroll
    for (int o = 8; o > 0; o >>= 1) mx = fmaxf(mx, __shfl_xor(mx, o, 16));
    float sum = 0.0f;
    for (int i = sub; i < TT; i += 16) { float e = __expf(Sr[i] - mx); Sr[i] = e; sum += e; }
#pragma unroll
    for (int o = 8; o > 0; o >>= 1) sum += __shfl_xor(sum, o, 16);
    float inv = 1.0f / sum;
    for (int i = sub; i < TT; i += 16) Sr[i] *= inv;
  }
  __syncthreads();

  // Phase 3: O = P @ V. Wave w -> n-tile (w&3), K-half (w>>2); halves combined in LDS.
  int nt = w & 3, ksec = w >> 2;
  v8f acc = {};
  const float*  Sa = S + n0 * TT + half * 16;                      // A row = lane&15
  const __bf16* vb = vp + (size_t)(nt * 16 + n0) * TT + half * 16; // B col = head dim
  {
    int kbeg = ksec * 1024;
#pragma unroll 4
    for (int k0 = kbeg; k0 < kbeg + 1024; k0 += 32) {
      v16bf B = *(const v16bf*)(vb + k0);
      const float* pr = Sa + k0;
      v16bf a;
#pragma unroll
      for (int i = 0; i < 8; i++) { a[i] = (__bf16)pr[i]; a[i + 8] = (__bf16)pr[i + 8]; }
      acc = wmma_bf16(a, B, acc);
    }
  }
  if (ksec == 1) {
#pragma unroll
    for (int r = 0; r < 8; r++) scratch[(r + half * 8) * 64 + nt * 16 + n0] = acc[r];
  }
  __syncthreads();
  if (ksec == 0) {
    int b = bh >> 4, h = bh & 15;
#pragma unroll
    for (int r = 0; r < 8; r++) {
      int m = r + half * 8;
      float v = acc[r] + scratch[m * 64 + nt * 16 + n0];
      ao[((size_t)(b * TT + t0 + m)) * CC + h * HD + nt * 16 + n0] = (__bf16)v;
    }
  }
}

// ---------------- output projection + bias ----------------------
// Per wave: 32x64 tile, manual ping-pong double buffer, f32 output.
__global__ void __launch_bounds__(256)
proj_gemm(const __bf16* __restrict__ ao, const __bf16* __restrict__ wp,
          const float* __restrict__ bp, float* __restrict__ out) {
  int wid  = blockIdx.x * 8 + (threadIdx.x >> 5);
  int mt   = wid >> 4;           // 256 m-tiles of 32 rows
  int ng   = wid & 15;
  int lane = threadIdx.x & 31;
  int n0   = lane & 15;
  int half = lane >> 4;
  int base_m = mt * 32, base_n = ng * 64;

  const __bf16* pa0 = ao + (size_t)(base_m + n0) * CC + half * 16;
  const __bf16* pa1 = ao + (size_t)(base_m + 16 + n0) * CC + half * 16;
  const __bf16* pw  = wp + (size_t)(base_n + n0) * CC + half * 16;

  v8f acc[8] = {};
  v16bf A0[2], A1[2], Bt[2][4];

  A0[0] = a_frag_bf16(pa0);
  A1[0] = a_frag_bf16(pa1);
#pragma unroll
  for (int j = 0; j < 4; j++) Bt[0][j] = *(const v16bf*)(pw + (size_t)j * 16 * CC);

  for (int k0 = 0; k0 < CC; k0 += 64) {
    A0[1] = a_frag_bf16(pa0 + k0 + 32);
    A1[1] = a_frag_bf16(pa1 + k0 + 32);
#pragma unroll
    for (int j = 0; j < 4; j++)
      Bt[1][j] = *(const v16bf*)(pw + (size_t)j * 16 * CC + k0 + 32);

#pragma unroll
    for (int j = 0; j < 4; j++) {
      acc[j]     = wmma_bf16(A0[0], Bt[0][j], acc[j]);
      acc[j + 4] = wmma_bf16(A1[0], Bt[0][j], acc[j + 4]);
    }

    if (k0 + 64 < CC) {
      A0[0] = a_frag_bf16(pa0 + k0 + 64);
      A1[0] = a_frag_bf16(pa1 + k0 + 64);
#pragma unroll
      for (int j = 0; j < 4; j++)
        Bt[0][j] = *(const v16bf*)(pw + (size_t)j * 16 * CC + k0 + 64);
    }

#pragma unroll
    for (int j = 0; j < 4; j++) {
      acc[j]     = wmma_bf16(A0[1], Bt[1][j], acc[j]);
      acc[j + 4] = wmma_bf16(A1[1], Bt[1][j], acc[j + 4]);
    }
  }

#pragma unroll
  for (int s2 = 0; s2 < 2; s2++) {
#pragma unroll
    for (int j = 0; j < 4; j++) {
      int d = base_n + j * 16 + n0;
      float bias = bp[d];
#pragma unroll
      for (int r = 0; r < 8; r++) {
        int m = base_m + s2 * 16 + r + half * 8;
        out[(size_t)m * CC + d] = acc[s2 * 4 + j][r] + bias;
      }
    }
  }
}

extern "C" void kernel_launch(void* const* d_in, const int* in_sizes, int n_in,
                              void* d_out, int out_size, void* d_ws, size_t ws_size,
                              hipStream_t stream) {
  const float* query = (const float*)d_in[0];
  const float* key_  = (const float*)d_in[1];
  const float* value = (const float*)d_in[2];
  const int*   mask  = (const int*)d_in[3];
  const float* Wq    = (const float*)d_in[4];
  const float* Wk    = (const float*)d_in[5];
  const float* Wv    = (const float*)d_in[6];
  const float* Wp    = (const float*)d_in[7];
  const float* bp    = (const float*)d_in[8];
  float* out = (float*)d_out;

  char* ws = (char*)d_ws;
  size_t off = 0;
  auto take = [&](size_t bytes) -> char* {
    char* p = ws + off;
    off += (bytes + 255) & ~(size_t)255;
    return p;
  };
  __bf16* wqb = (__bf16*)take((size_t)CC * CC * 2);
  __bf16* wkb = (__bf16*)take((size_t)CC * CC * 2);
  __bf16* wvb = (__bf16*)take((size_t)CC * CC * 2);
  __bf16* wpb = (__bf16*)take((size_t)CC * CC * 2);
  __bf16* qh  = (__bf16*)take((size_t)BB * HH * TT * HD * 2);
  __bf16* kh  = (__bf16*)take((size_t)BB * HH * TT * HD * 2);
  __bf16* vT  = (__bf16*)take((size_t)BB * HH * HD * TT * 2);
  __bf16* ao  = (__bf16*)take((size_t)BB * TT * CC * 2);

  // 1) weights -> bf16
  cvt_f32_bf16<<<1024, 256, 0, stream>>>(Wq, wqb, CC * CC);
  cvt_f32_bf16<<<1024, 256, 0, stream>>>(Wk, wkb, CC * CC);
  cvt_f32_bf16<<<1024, 256, 0, stream>>>(Wv, wvb, CC * CC);
  cvt_f32_bf16<<<1024, 256, 0, stream>>>(Wp, wpb, CC * CC);

  // 2) fused QKV projection: 3 * 256 m-tiles * 16 n-groups = 12288 wave-tiles / 8 waves
  qkv_gemm<<<1536, 256, 0, stream>>>(query, key_, value, wqb, wkb, wvb, qh, kh, vT);

  // 3) attention: B*H*(T/16) = 8192 blocks, ~134 KB dynamic LDS (S + scratch + q tile)
  size_t shmem = (size_t)(16 * TT + 16 * 64) * sizeof(float) + (size_t)16 * 64 * sizeof(__bf16);
  (void)hipFuncSetAttribute((const void*)attn_kernel,
                            hipFuncAttributeMaxDynamicSharedMemorySize, (int)shmem);
  attn_kernel<<<8192, 256, shmem, stream>>>(qh, kh, vT, mask, ao);

  // 4) output projection + bias: 256*16 = 4096 wave-tiles / 8 waves
  proj_gemm<<<512, 256, 0, stream>>>(ao, wpb, bp, out);
}
